// HiddenLSTMEncoderDecoder_71159018160177
// MI455X (gfx1250) — matrix operand order repaired and checked by reading, not compiled, not run on hardware
//
#include <hip/hip_runtime.h>
#include <hip/hip_bf16.h>

// Problem dims
#define B_    64
#define T_    512
#define F_    51
#define H_    512
#define G4H   2048
#define MROWS (B_ * T_)   // 32768
#define KPAD0 64          // padded input dim for F=51 layers

typedef __bf16 bf16;
typedef __attribute__((ext_vector_type(16))) __bf16 v16bf;
typedef __attribute__((ext_vector_type(8)))  float  v8f;
typedef __attribute__((ext_vector_type(4)))  int    v4i_t;

// ---------------------------------------------------------------- CDNA5 async global->LDS

#if __has_builtin(__builtin_amdgcn_global_load_async_to_lds_b128)
#define HAVE_ASYNC_LDS 1
#else
#define HAVE_ASYNC_LDS 0
#endif

__device__ __forceinline__ void wait_async0() {
#if __has_builtin(__builtin_amdgcn_s_wait_asynccnt)
  __builtin_amdgcn_s_wait_asynccnt(0);
#else
  asm volatile("s_wait_asynccnt 0" ::: "memory");
#endif
}

#if HAVE_ASYNC_LDS
__device__ __forceinline__ void async_b128(const void* g, void* l) {
  __builtin_amdgcn_global_load_async_to_lds_b128(
      (__attribute__((address_space(1))) v4i_t*)(void*)g,
      (__attribute__((address_space(3))) v4i_t*)l, 0, 0);
}
#endif

// ---------------------------------------------------------------- WMMA helpers

__device__ __forceinline__ v8f wmma_bf16(v16bf a, v16bf b, v8f c) {
  // D = A(16x32) * B(32x16) + C, f32 accumulate
  return __builtin_amdgcn_wmma_f32_16x16x32_bf16(
      /*neg_a=*/false, a, /*neg_b=*/false, b,
      /*c_mod=*/(short)0, c, /*reuse_a=*/false, /*reuse_b=*/false);
}

// A fragment: 16x32 tile at (m0,k0) of row-major [rows x ld] bf16 matrix.
// Lane l holds row m0+(l&15); lanes 0-15 cover K {0..7,16..23}, lanes 16-31 {8..15,24..31}.
__device__ __forceinline__ v16bf load_frag_a(const bf16* src, int ld, int m0, int k0) {
  int lane = threadIdx.x & 31;
  const bf16* p = src + (size_t)(m0 + (lane & 15)) * ld + k0 + ((lane >> 4) << 3);
  v16bf a;
#pragma unroll
  for (int r = 0; r < 8; ++r) {
    int k = ((r & 3) << 1) + ((r >> 2) << 4);
    a[2 * r]     = p[k];
    a[2 * r + 1] = p[k + 1];
  }
  return a;
}

// A fragment with decoder time-shift: matrix row r=(b,t) reads memory row (b, (t-1) mod T).
__device__ __forceinline__ v16bf load_frag_a_shift(const bf16* src, int ld, int m0, int k0) {
  int lane = threadIdx.x & 31;
  int r0 = m0 + (lane & 15);
  int b  = r0 >> 9;                 // T_ == 512
  int t  = (r0 + T_ - 1) & (T_ - 1);
  const bf16* p = src + (size_t)((b << 9) | t) * ld + k0 + ((lane >> 4) << 3);
  v16bf a;
#pragma unroll
  for (int r = 0; r < 8; ++r) {
    int k = ((r & 3) << 1) + ((r >> 2) << 4);
    a[2 * r]     = p[k];
    a[2 * r + 1] = p[k + 1];
  }
  return a;
}

// B fragment: B[k][n] = w[n0+n][k0+k], w row-major [N x ld].
// Lane l owns column n0+(l&15); lanes 0-15 cover K {0..7,16..23}, lanes 16-31 {8..15,24..31}.
__device__ __forceinline__ v16bf load_frag_b(const bf16* w, int ld, int n0, int k0) {
  int lane = threadIdx.x & 31;
  const bf16* p = w + (size_t)(n0 + (lane & 15)) * ld + k0 + ((lane >> 4) << 3);
  v16bf b;
#pragma unroll
  for (int r = 0; r < 8; ++r) { int k = ((r & 3) << 1) + ((r >> 2) << 4); b[2 * r] = p[k]; b[2 * r + 1] = p[k + 1]; }
  return b;
}

__device__ __forceinline__ float sigm(float x) { return 1.f / (1.f + __expf(-x)); }
__device__ __forceinline__ float tanh_f(float x) {
  float cx = fminf(fmaxf(x, -15.f), 15.f);
  float e = __expf(2.f * cx);
  return (e - 1.f) / (e + 1.f);
}

// ---------------------------------------------------------------- utility kernels

__global__ void zero_kernel(int* p, int n) {
  int i = blockIdx.x * blockDim.x + threadIdx.x;
  if (i < n) p[i] = 0;
}

// fp32 [srows x scols] -> bf16 [drows x dcols], zero-padded
__global__ void pad_cvt_kernel(const float* __restrict__ src, bf16* __restrict__ dst,
                               int srows, int scols, int drows, int dcols) {
  int i = blockIdx.x * blockDim.x + threadIdx.x;
  int n = drows * dcols;
  if (i >= n) return;
  int r = i / dcols, c = i - r * dcols;
  float v = (r < srows && c < scols) ? src[(size_t)r * scols + c] : 0.f;
  dst[i] = (bf16)v;
}

// ---------------------------------------------------------------- xg = in @ w_ih^T + (b_ih + b_hh)
// in: bf16 [MROWS x ld]; w: bf16 [2048 x ld]; xg: f32 [MROWS x 2048]
// grid (MROWS/32, 2048/64), 256 threads: 8 waves, each one 16x16 tile (2 m-tiles x 4 n-tiles).

__global__ void gemm_xg_kernel(const bf16* __restrict__ in,
                               const bf16* __restrict__ w, int ld,
                               const float* __restrict__ b_ih,
                               const float* __restrict__ b_hh,
                               float* __restrict__ xg, int shifted) {
  int wave = threadIdx.x >> 5;
  int lane = threadIdx.x & 31;
  int mt = wave & 1, nt = wave >> 1;
  int m0 = blockIdx.x * 32 + mt * 16;
  int n0 = blockIdx.y * 64 + nt * 16;
  v8f acc = {};
  int nk = ld >> 5;
  for (int kc = 0; kc < nk; ++kc) {
    int k0 = kc << 5;
    v16bf a = shifted ? load_frag_a_shift(in, ld, m0, k0)
                      : load_frag_a(in, ld, m0, k0);
    v16bf b = load_frag_b(w, ld, n0, k0);
    acc = wmma_bf16(a, b, acc);
  }
  int n = n0 + (lane & 15);
  float bias = b_ih[n] + b_hh[n];
#pragma unroll
  for (int r = 0; r < 8; ++r) {
    int m = m0 + r + ((lane >> 4) << 3);
    xg[(size_t)m * G4H + n] = acc[r] + bias;
  }
}

// ---------------------------------------------------------------- recurrent scan (persistent)
// 32 blocks x 256 threads. Block jb owns hidden cols [jb*16, jb*16+16) of all 4 gates.
// Dyn LDS: 64KB w_hh slice + 64KB h + 16KB gate staging = 144KB (needs CDNA5's 320KB WGP LDS).

#define SCAN_BLOCKS 32
#define SCAN_LDS    (65536 + 65536 + 16384)

__global__ void lstm_scan_kernel(const float* __restrict__ xg,
                                 const bf16* __restrict__ whh,   // [2048 x 512] bf16
                                 bf16* __restrict__ hbuf0, bf16* __restrict__ hbuf1,
                                 float* __restrict__ cstate,     // [64 x 512] f32 (in/out)
                                 bf16* __restrict__ ys,          // [MROWS x 512] bf16
                                 int* __restrict__ cnt) {
  extern __shared__ char smem[];
  bf16*  lds_w = (bf16*)smem;                    // [64 rows (4g x 16j)] x 512
  bf16*  lds_h = (bf16*)(smem + 65536);          // [64 b] x 512
  float* lds_g = (float*)(smem + 131072);        // [4 g][64 b][16 j]

  const int tid  = threadIdx.x;
  const int lane = tid & 31;
  const int wave = tid >> 5;
  const int j0   = blockIdx.x * 16;

  // Preload this block's w_hh slice: lds row (g*16+jj) <- whh row (g*512 + j0 + jj).
  // Uses the gfx1250 async global->LDS engine (ASYNCcnt) when available.
  {
#if HAVE_ASYNC_LDS
    for (int v = tid; v < 4096; v += 256) {
      int row = v >> 6, q = v & 63;
      int g = row >> 4, jj = row & 15;
      const uint4* src = (const uint4*)(whh + (size_t)(g * H_ + j0 + jj) * H_) + q;
      async_b128(src, (uint4*)lds_w + v);
    }
    wait_async0();
#else
    uint4* dst = (uint4*)lds_w;
    for (int v = tid; v < 4096; v += 256) {
      int row = v >> 6, q = v & 63;
      int g = row >> 4, jj = row & 15;
      const uint4* src = (const uint4*)(whh + (size_t)(g * H_ + j0 + jj) * H_);
      dst[v] = src[q];
    }
#endif
  }

  // Register-resident cell state: element e = q*256+tid -> b=e>>4, jj=e&15
  float creg[4];
#pragma unroll
  for (int q = 0; q < 4; ++q) {
    int e = q * 256 + tid;
    creg[q] = cstate[(size_t)(e >> 4) * H_ + j0 + (e & 15)];
  }
  __syncthreads();

  const int mt    = wave & 3;          // batch tile (4 x 16 = 64)
  const int gbase = (wave >> 2) << 1;  // gate pair {0,1} or {2,3}

  for (int t = 0; t < T_; ++t) {
    const bf16* hr = (t & 1) ? hbuf1 : hbuf0;   // double-buffered h exchange
    bf16*       hw = (t & 1) ? hbuf0 : hbuf1;

    { // stage h[64x512] into LDS (64KB) via async DMA when available
#if HAVE_ASYNC_LDS
      const uint4* src = (const uint4*)hr;
      for (int v = tid; v < 4096; v += 256)
        async_b128(src + v, (uint4*)lds_h + v);
      wait_async0();
#else
      uint4* dst = (uint4*)lds_h;
      const uint4* src = (const uint4*)hr;
      for (int v = tid; v < 4096; v += 256) dst[v] = src[v];
#endif
    }
    __syncthreads();

    v8f acc0 = {}; v8f acc1 = {};
#pragma unroll 4
    for (int kc = 0; kc < 16; ++kc) {
      int k0 = kc << 5;
      v16bf a  = load_frag_a(lds_h, H_, mt * 16, k0);
      v16bf b0 = load_frag_b(lds_w, H_, gbase * 16, k0);
      v16bf b1 = load_frag_b(lds_w, H_, (gbase + 1) * 16, k0);
      acc0 = wmma_bf16(a, b0, acc0);
      acc1 = wmma_bf16(a, b1, acc1);
    }
#pragma unroll
    for (int r = 0; r < 8; ++r) {
      int m  = mt * 16 + r + ((lane >> 4) << 3);
      int jj = lane & 15;
      lds_g[gbase * 1024 + m * 16 + jj]       = acc0[r];
      lds_g[(gbase + 1) * 1024 + m * 16 + jj] = acc1[r];
    }
    __syncthreads();

    // Gates + state update (PyTorch order i,f,g,o)
#pragma unroll
    for (int q = 0; q < 4; ++q) {
      int e = q * 256 + tid;
      int b = e >> 4, jj = e & 15;
      size_t base = (size_t)(b * T_ + t) * G4H + j0 + jj;
      float iv = lds_g[0 * 1024 + b * 16 + jj] + xg[base];
      float fv = lds_g[1 * 1024 + b * 16 + jj] + xg[base + 512];
      float gv = lds_g[2 * 1024 + b * 16 + jj] + xg[base + 1024];
      float ov = lds_g[3 * 1024 + b * 16 + jj] + xg[base + 1536];
      float c = sigm(fv) * creg[q] + sigm(iv) * tanh_f(gv);
      float h = sigm(ov) * tanh_f(c);
      creg[q] = c;
      bf16 hb = (bf16)h;
      hw[(size_t)b * H_ + j0 + jj] = hb;
      ys[(size_t)(b * T_ + t) * H_ + j0 + jj] = hb;
      // Prefetch next timestep's gate rows across the grid barrier (global_prefetch).
      if (t + 1 < T_) {
        __builtin_prefetch(&xg[base + G4H], 0, 1);
        __builtin_prefetch(&xg[base + G4H + 512], 0, 1);
        __builtin_prefetch(&xg[base + G4H + 1024], 0, 1);
        __builtin_prefetch(&xg[base + G4H + 1536], 0, 1);
      }
    }

    // Grid-wide barrier (monotonic counter; h is double-buffered so one barrier/step)
    __threadfence();
    __syncthreads();
    if (tid == 0) {
      __hip_atomic_fetch_add(cnt, 1, __ATOMIC_RELEASE, __HIP_MEMORY_SCOPE_AGENT);
      int target = (t + 1) * (int)gridDim.x;
      while (__hip_atomic_load(cnt, __ATOMIC_ACQUIRE, __HIP_MEMORY_SCOPE_AGENT) < target)
        __builtin_amdgcn_s_sleep(1);
    }
    __syncthreads();
    __threadfence();
  }

  // Persist final c (decoder init / unused for final layers)
#pragma unroll
  for (int q = 0; q < 4; ++q) {
    int e = q * 256 + tid;
    cstate[(size_t)(e >> 4) * H_ + j0 + (e & 15)] = creg[q];
  }
}

// ---------------------------------------------------------------- out = d1 @ out_w^T + out_b
// d1: bf16 [MROWS x 512]; wo: bf16 [64 x 512] (rows >= 51 zero); out: f32 [MROWS x 51]
// grid (256, 4), 256 threads: 8 waves, each one m-tile.

__global__ void proj_kernel(const bf16* __restrict__ d1, const bf16* __restrict__ wo,
                            const float* __restrict__ bo, float* __restrict__ out) {
  int wave = threadIdx.x >> 5;
  int lane = threadIdx.x & 31;
  int m0 = (blockIdx.x * 8 + wave) * 16;
  int n0 = blockIdx.y * 16;
  v8f acc = {};
  for (int kc = 0; kc < 16; ++kc) {
    int k0 = kc << 5;
    v16bf a = load_frag_a(d1, H_, m0, k0);
    v16bf b = load_frag_b(wo, H_, n0, k0);
    acc = wmma_bf16(a, b, acc);
  }
  int f = n0 + (lane & 15);
  if (f < F_) {
    float bias = bo[f];
#pragma unroll
    for (int r = 0; r < 8; ++r) {
      int m = m0 + r + ((lane >> 4) << 3);
      out[(size_t)m * F_ + f] = acc[r] + bias;
    }
  }
}

// ---------------------------------------------------------------- workspace layout

static constexpr size_t OFF_XG   = 0;                          // 32768*2048*4 = 256MB
static constexpr size_t OFF_XBF  = OFF_XG  + 268435456ull;     // 32768*64*2
static constexpr size_t OFF_YSA  = OFF_XBF + 4194304ull;       // 32768*512*2
static constexpr size_t OFF_YSB  = OFF_YSA + 33554432ull;
static constexpr size_t OFF_WHH  = OFF_YSB + 33554432ull;      // 4 x (2048*512*2)
static constexpr size_t OFF_WIH0 = OFF_WHH  + 4ull * 2097152ull; // enc0: 2048*64*2
static constexpr size_t OFF_WIH1 = OFF_WIH0 + 262144ull;         // enc1: 2048*512*2
static constexpr size_t OFF_WIH2 = OFF_WIH1 + 2097152ull;        // dec0
static constexpr size_t OFF_WIH3 = OFF_WIH2 + 262144ull;         // dec1
static constexpr size_t OFF_WO   = OFF_WIH3 + 2097152ull;        // 64*512*2
static constexpr size_t OFF_C0   = OFF_WO   + 65536ull;          // 32768 f32
static constexpr size_t OFF_C1   = OFF_C0   + 131072ull;
static constexpr size_t OFF_H0A  = OFF_C1   + 131072ull;         // 32768 bf16
static constexpr size_t OFF_H0B  = OFF_H0A  + 65536ull;
static constexpr size_t OFF_H1A  = OFF_H0B  + 65536ull;
static constexpr size_t OFF_H1B  = OFF_H1A  + 65536ull;
static constexpr size_t OFF_CNT  = OFF_H1B  + 65536ull;          // 4 ints

extern "C" void kernel_launch(void* const* d_in, const int* in_sizes, int n_in,
                              void* d_out, int out_size, void* d_ws, size_t ws_size,
                              hipStream_t stream) {
  (void)in_sizes; (void)n_in; (void)out_size; (void)ws_size;
  char* ws = (char*)d_ws;

  const float* x = (const float*)d_in[0];
  const float* w_ih[4] = { (const float*)d_in[1], (const float*)d_in[5],
                           (const float*)d_in[9], (const float*)d_in[13] };
  const float* w_hh[4] = { (const float*)d_in[2], (const float*)d_in[6],
                           (const float*)d_in[10], (const float*)d_in[14] };
  const float* b_ih[4] = { (const float*)d_in[3], (const float*)d_in[7],
                           (const float*)d_in[11], (const float*)d_in[15] };
  const float* b_hh[4] = { (const float*)d_in[4], (const float*)d_in[8],
                           (const float*)d_in[12], (const float*)d_in[16] };
  const float* out_w = (const float*)d_in[17];
  const float* out_b = (const float*)d_in[18];

  float* xg  = (float*)(ws + OFF_XG);
  bf16*  xbf = (bf16*)(ws + OFF_XBF);
  bf16*  ysA = (bf16*)(ws + OFF_YSA);
  bf16*  ysB = (bf16*)(ws + OFF_YSB);
  bf16*  whhb[4] = { (bf16*)(ws + OFF_WHH), (bf16*)(ws + OFF_WHH + 2097152ull),
                     (bf16*)(ws + OFF_WHH + 2ull * 2097152ull),
                     (bf16*)(ws + OFF_WHH + 3ull * 2097152ull) };
  bf16*  wihb[4] = { (bf16*)(ws + OFF_WIH0), (bf16*)(ws + OFF_WIH1),
                     (bf16*)(ws + OFF_WIH2), (bf16*)(ws + OFF_WIH3) };
  bf16*  wob = (bf16*)(ws + OFF_WO);
  float* c0  = (float*)(ws + OFF_C0);
  float* c1  = (float*)(ws + OFF_C1);
  bf16*  h0a = (bf16*)(ws + OFF_H0A);
  bf16*  h0b = (bf16*)(ws + OFF_H0B);
  bf16*  h1a = (bf16*)(ws + OFF_H1A);
  bf16*  h1b = (bf16*)(ws + OFF_H1B);
  int*   cnt = (int*)(ws + OFF_CNT);

  // 1) zero state + barrier counters (524304 bytes = 131076 ints)
  {
    int nz = 131076;
    zero_kernel<<<(nz + 255) / 256, 256, 0, stream>>>((int*)(ws + OFF_C0), nz);
  }

  // 2) convert/pad weights + inputs to bf16
  auto cvt = [&](const float* s, bf16* d, int sr, int sc, int dr, int dc) {
    int n = dr * dc;
    pad_cvt_kernel<<<(n + 255) / 256, 256, 0, stream>>>(s, d, sr, sc, dr, dc);
  };
  cvt(x, xbf, MROWS, F_, MROWS, KPAD0);
  cvt(w_ih[0], wihb[0], G4H, F_,  G4H, KPAD0);
  cvt(w_ih[1], wihb[1], G4H, H_,  G4H, H_);
  cvt(w_ih[2], wihb[2], G4H, F_,  G4H, KPAD0);
  cvt(w_ih[3], wihb[3], G4H, H_,  G4H, H_);
  for (int l = 0; l < 4; ++l) cvt(w_hh[l], whhb[l], G4H, H_, G4H, H_);
  cvt(out_w, wob, F_, H_, 64, H_);

  dim3 gg(MROWS / 32, G4H / 64);

  // 3) encoder layer 0
  gemm_xg_kernel<<<gg, 256, 0, stream>>>(xbf, wihb[0], KPAD0, b_ih[0], b_hh[0], xg, 0);
  lstm_scan_kernel<<<SCAN_BLOCKS, 256, SCAN_LDS, stream>>>(xg, whhb[0], h0a, h0b, c0, ysA, cnt + 0);
  // encoder layer 1
  gemm_xg_kernel<<<gg, 256, 0, stream>>>(ysA, wihb[1], H_, b_ih[1], b_hh[1], xg, 0);
  lstm_scan_kernel<<<SCAN_BLOCKS, 256, SCAN_LDS, stream>>>(xg, whhb[1], h1a, h1b, c1, ysB, cnt + 1);
  // decoder layer 0 (shifted input; init state = encoder layer-0 finals left in h0a/c0)
  gemm_xg_kernel<<<gg, 256, 0, stream>>>(xbf, wihb[2], KPAD0, b_ih[2], b_hh[2], xg, 1);
  lstm_scan_kernel<<<SCAN_BLOCKS, 256, SCAN_LDS, stream>>>(xg, whhb[2], h0a, h0b, c0, ysA, cnt + 2);
  // decoder layer 1 (init state = encoder layer-1 finals in h1a/c1)
  gemm_xg_kernel<<<gg, 256, 0, stream>>>(ysA, wihb[3], H_, b_ih[3], b_hh[3], xg, 0);
  lstm_scan_kernel<<<SCAN_BLOCKS, 256, SCAN_LDS, stream>>>(xg, whhb[3], h1a, h1b, c1, ysB, cnt + 3);

  // 4) output projection
  proj_kernel<<<dim3(MROWS / 128, 4), 256, 0, stream>>>(ysB, wob, out_b, (float*)d_out);
}
